// FrenchToEnglish_9732395893024
// MI455X (gfx1250) — matrix operand order repaired and checked by reading, not compiled, not run on hardware
//
#include <hip/hip_runtime.h>
#include <hip/hip_bf16.h>
#include <stdint.h>

// Model dims
#define Hh   256
#define Ee   300
#define Bz   64
#define Sz   64
#define Tz   32
#define VEv  32000
#define EPAD 320     // E padded to mult of 32 (and 64)
#define CPAD 832     // (2H+E)=812 padded to mult of 32 (and 64)

typedef __attribute__((ext_vector_type(16))) __bf16 v16bf;
typedef __attribute__((ext_vector_type(8)))  float  v8f;
typedef unsigned short u16;

struct U128 { unsigned int a, b, c, d; };
union Frag16 { U128 u[2]; v16bf v; };

__device__ __forceinline__ u16 f2bf(float f) {
    unsigned int u = __float_as_uint(f);
    unsigned int r = (u + 0x7FFFu + ((u >> 16) & 1u)) >> 16;
    return (u16)r;
}

__device__ __forceinline__ void load_a(Frag16& f, const u16* p, int kk) {
    f.u[0] = *(const U128*)(p + kk);
    f.u[1] = *(const U128*)(p + kk + 16);
}
__device__ __forceinline__ void load_b(Frag16& f, const u16* p, int kk) {
    f.u[0] = *(const U128*)(p + kk);
    f.u[1] = *(const U128*)(p + kk + 16);
}

#define WMMA(ACC, FA, FB) \
    ACC = __builtin_amdgcn_wmma_f32_16x16x32_bf16(false, (FA).v, false, (FB).v, (short)0, ACC, false, false)

// ---------------------------------------------------------------------------
// GEMM: C[M,N] = A[M,K](bf16) * W[N,K](bf16)^T + bias[N]
// Block: 64 rows (4 M-tiles) x 256 cols (8 waves x 2 N-tiles); 8 wmma/K-step.
// K unrolled by 2 with disjoint fragment register sets (double buffer) so
// loads for kk+32 fly while wmmas for kk execute.  K mult of 64; M mult of 64.
// ---------------------------------------------------------------------------
__global__ __launch_bounds__(256) void k_gemm4(
    const u16* __restrict__ A, int lda,
    const u16* __restrict__ W, int ldb,
    float* __restrict__ C, int ldc,
    u16* __restrict__ Cb, int ldcb,
    const float* __restrict__ bias,
    int N, int K)
{
    const int lane = threadIdx.x & 31;
    const int wave = threadIdx.x >> 5;
    const int m0 = blockIdx.y * 64;
    const int n0 = (blockIdx.x * 8 + wave) * 32;
    if (n0 >= N) return;                       // wave-uniform

    const int kse = (lane >> 4) & 1;
    const int l15 = lane & 15;

    const u16* ap0 = A + (size_t)(m0 +  0 + l15) * lda + 8 * kse;
    const u16* ap1 = A + (size_t)(m0 + 16 + l15) * lda + 8 * kse;
    const u16* ap2 = A + (size_t)(m0 + 32 + l15) * lda + 8 * kse;
    const u16* ap3 = A + (size_t)(m0 + 48 + l15) * lda + 8 * kse;
    int brow0 = n0 + l15;      if (brow0 >= N) brow0 = 0;
    int brow1 = n0 + 16 + l15; if (brow1 >= N) brow1 = 0;
    const u16* bp0 = W + (size_t)brow0 * ldb + 8 * kse;
    const u16* bp1 = W + (size_t)brow1 * ldb + 8 * kse;

    v8f acc[8];
#pragma unroll
    for (int i = 0; i < 8; ++i) acc[i] = (v8f){0.f,0.f,0.f,0.f,0.f,0.f,0.f,0.f};

    // double-buffered fragments (disjoint register sets, no copies)
    Frag16 ca0, ca1, ca2, ca3, cb0, cb1;
    Frag16 na0, na1, na2, na3, nb0, nb1;
    load_b(cb0, bp0, 0); load_b(cb1, bp1, 0);
    load_a(ca0, ap0, 0); load_a(ca1, ap1, 0);
    load_a(ca2, ap2, 0); load_a(ca3, ap3, 0);

    for (int kk = 0; kk < K; kk += 64) {
        // loads for kk+32 issue before wmmas on kk (K is a multiple of 64)
        load_b(nb0, bp0, kk + 32); load_b(nb1, bp1, kk + 32);
        load_a(na0, ap0, kk + 32); load_a(na1, ap1, kk + 32);
        load_a(na2, ap2, kk + 32); load_a(na3, ap3, kk + 32);
        __builtin_prefetch((const void*)(bp0 + kk + 96), 0, 1);   // speculative: OOB dropped
        __builtin_prefetch((const void*)(bp1 + kk + 96), 0, 1);

        WMMA(acc[0], ca0, cb0); WMMA(acc[1], ca0, cb1);
        WMMA(acc[2], ca1, cb0); WMMA(acc[3], ca1, cb1);
        WMMA(acc[4], ca2, cb0); WMMA(acc[5], ca2, cb1);
        WMMA(acc[6], ca3, cb0); WMMA(acc[7], ca3, cb1);

        if (kk + 64 < K) {
            load_b(cb0, bp0, kk + 64); load_b(cb1, bp1, kk + 64);
            load_a(ca0, ap0, kk + 64); load_a(ca1, ap1, kk + 64);
            load_a(ca2, ap2, kk + 64); load_a(ca3, ap3, kk + 64);
        }
        WMMA(acc[0], na0, nb0); WMMA(acc[1], na0, nb1);
        WMMA(acc[2], na1, nb0); WMMA(acc[3], na1, nb1);
        WMMA(acc[4], na2, nb0); WMMA(acc[5], na2, nb1);
        WMMA(acc[6], na3, nb0); WMMA(acc[7], na3, nb1);
    }

    const int nA = n0 + l15;
    const int nB = n0 + 16 + l15;
    const bool okA = nA < N, okB = nB < N;
    float bA = 0.f, bB = 0.f;
    if (bias) { if (okA) bA = bias[nA]; if (okB) bB = bias[nB]; }
#pragma unroll
    for (int i = 0; i < 4; ++i) {
        const int mb = m0 + 16 * i + 8 * kse;
#pragma unroll
        for (int v = 0; v < 8; ++v) {
            const int m = mb + v;
            const float xA = acc[2 * i][v] + bA;
            const float xB = acc[2 * i + 1][v] + bB;
            if (okA) {
                if (C)  C[(size_t)m * ldc + nA] = xA;
                if (Cb) Cb[(size_t)m * ldcb + nA] = f2bf(xA);
            }
            if (okB) {
                if (C)  C[(size_t)m * ldc + nB] = xB;
                if (Cb) Cb[(size_t)m * ldcb + nB] = f2bf(xB);
            }
        }
    }
}

// ---------------------------------------------------------------------------
// Persistent GRU scan: one workgroup (32 waves), h resident in LDS.
// Per step: gh = h_bf @ Whh^T + bhh (WMMA, A from LDS) -> LDS; barrier;
//           fused GRU pointwise updating LDS h and writing y; barrier.
// LDS: gh 64x768 f32 (192KB) + h 64x256 f32 (64KB) + h bf16 (32KB) = 288KB.
// ---------------------------------------------------------------------------
__global__ __launch_bounds__(1024) void k_scan(
    const float* __restrict__ gi,      // [B*S, 768]  row r = b*S + t
    const u16*   __restrict__ Whh,     // [768, 256] bf16
    const float* __restrict__ bhh,     // [768]
    float*       __restrict__ yf,      // nullable, row r=(b*S+t), stride 512 (col off applied)
    u16*         __restrict__ ybf,     // nullable, same layout
    float*       __restrict__ hs_out,  // [64*256] final hidden
    int reverse)
{
    extern __shared__ char smem[];
    float* sh_gh  = (float*)smem;                         // 64*768
    float* sh_hf  = (float*)(smem + 64 * 768 * 4);        // 64*256
    u16*   sh_hbf = (u16*)  (smem + 64 * 768 * 4 + 64 * 256 * 4);

    const int tid = threadIdx.x;
    const int lane = tid & 31;
    const int wave = tid >> 5;
    const int kse = (lane >> 4) & 1;
    const int l15 = lane & 15;

    for (int i = tid; i < 64 * 256; i += 1024) { sh_hf[i] = 0.f; sh_hbf[i] = 0; }
    __syncthreads();

    for (int step = 0; step < Sz; ++step) {
        const int t = reverse ? (Sz - 1 - step) : step;

        // ---- GEMM phase: 4 M-tiles x 24 N-pairs = 96 units, 3 per wave ----
        for (int p = wave; p < 96; p += 32) {
            const int m0 = (p & 3) * 16;
            const int n0 = (p >> 2) * 32;
            const u16* arow = sh_hbf + (m0 + l15) * 256 + 8 * kse;
            const u16* bp0 = Whh + (size_t)(n0 + l15) * 256 + 8 * kse;
            const u16* bp1 = Whh + (size_t)(n0 + 16 + l15) * 256 + 8 * kse;
            v8f acc0 = {0.f,0.f,0.f,0.f,0.f,0.f,0.f,0.f};
            v8f acc1 = {0.f,0.f,0.f,0.f,0.f,0.f,0.f,0.f};
#pragma unroll
            for (int kk = 0; kk < 256; kk += 32) {
                Frag16 fa, fb0, fb1;
                load_a(fa, arow, kk);
                load_b(fb0, bp0, kk);
                load_b(fb1, bp1, kk);
                WMMA(acc0, fa, fb0);
                WMMA(acc1, fa, fb1);
            }
            const int nA = n0 + l15, nB = n0 + 16 + l15;
            const float bA = bhh[nA], bB = bhh[nB];
            const int mb = m0 + 8 * kse;
#pragma unroll
            for (int v = 0; v < 8; ++v) {
                sh_gh[(mb + v) * 768 + nA] = acc0[v] + bA;
                sh_gh[(mb + v) * 768 + nB] = acc1[v] + bB;
            }
        }
        __syncthreads();

        // ---- pointwise GRU ----
        for (int i = tid; i < 64 * 256; i += 1024) {
            const int b = i >> 8, j = i & 255;
            const float* gib = gi + ((size_t)b * Sz + t) * 768;
            const float ir = gib[j], iz = gib[256 + j], in = gib[512 + j];
            const float* ghb = sh_gh + b * 768;
            const float hr = ghb[j], hz = ghb[256 + j], hn = ghb[512 + j];
            const float h = sh_hf[i];
            const float r = 1.f / (1.f + __expf(-(ir + hr)));
            const float z = 1.f / (1.f + __expf(-(iz + hz)));
            const float n = tanhf(in + r * hn);
            const float ho = (1.f - z) * n + z * h;
            sh_hf[i] = ho;
            sh_hbf[i] = f2bf(ho);
            const size_t yr = ((size_t)b * Sz + t) * 512 + j;
            if (yf)  yf[yr] = ho;
            if (ybf) ybf[yr] = f2bf(ho);
        }
        __syncthreads();
    }

    for (int i = tid; i < 64 * 256; i += 1024) hs_out[i] = sh_hf[i];
}

// ---------------------------------------------------------------------------
// Fused decoder GRU cell: gh GEMM (WMMA) into LDS + pointwise update.
// h may update in place (barrier separates read and write phases).
// ---------------------------------------------------------------------------
__global__ __launch_bounds__(1024) void k_grustep(
    const float* __restrict__ gi,        // [64, 768]
    const u16*   __restrict__ hprev_bf,  // [64, 256]
    const u16*   __restrict__ Whh,       // [768, 256]
    const float* __restrict__ bhh,
    const float* __restrict__ hprev_f,   // [64, 256]
    float*       __restrict__ hnew_f,
    u16*         __restrict__ hnew_bf)
{
    extern __shared__ char smem[];
    float* sh_gh = (float*)smem;         // 64*768
    const int tid = threadIdx.x;
    const int lane = tid & 31;
    const int wave = tid >> 5;
    const int kse = (lane >> 4) & 1;
    const int l15 = lane & 15;

    for (int p = wave; p < 96; p += 32) {
        const int m0 = (p & 3) * 16;
        const int n0 = (p >> 2) * 32;
        const u16* arow = hprev_bf + (size_t)(m0 + l15) * 256 + 8 * kse;
        const u16* bp0 = Whh + (size_t)(n0 + l15) * 256 + 8 * kse;
        const u16* bp1 = Whh + (size_t)(n0 + 16 + l15) * 256 + 8 * kse;
        v8f acc0 = {0.f,0.f,0.f,0.f,0.f,0.f,0.f,0.f};
        v8f acc1 = {0.f,0.f,0.f,0.f,0.f,0.f,0.f,0.f};
#pragma unroll
        for (int kk = 0; kk < 256; kk += 32) {
            Frag16 fa, fb0, fb1;
            load_a(fa, arow, kk);
            load_b(fb0, bp0, kk);
            load_b(fb1, bp1, kk);
            WMMA(acc0, fa, fb0);
            WMMA(acc1, fa, fb1);
        }
        const int nA = n0 + l15, nB = n0 + 16 + l15;
        const float bA = bhh[nA], bB = bhh[nB];
        const int mb = m0 + 8 * kse;
#pragma unroll
        for (int v = 0; v < 8; ++v) {
            sh_gh[(mb + v) * 768 + nA] = acc0[v] + bA;
            sh_gh[(mb + v) * 768 + nB] = acc1[v] + bB;
        }
    }
    __syncthreads();

    for (int i = tid; i < 64 * 256; i += 1024) {
        const int b = i >> 8, j = i & 255;
        const float* gib = gi + (size_t)b * 768;
        const float ir = gib[j], iz = gib[256 + j], in = gib[512 + j];
        const float* ghb = sh_gh + b * 768;
        const float hr = ghb[j], hz = ghb[256 + j], hn = ghb[512 + j];
        const float h = hprev_f[i];
        const float r = 1.f / (1.f + __expf(-(ir + hr)));
        const float z = 1.f / (1.f + __expf(-(iz + hz)));
        const float n = tanhf(in + r * hn);
        const float ho = (1.f - z) * n + z * h;
        hnew_f[i] = ho;
        hnew_bf[i] = f2bf(ho);
    }
}

// ---------------------------------------------------------------------------
__global__ void k_cvt(const float* __restrict__ s, u16* __restrict__ d,
                      int rows, int Kin, int Kpad)
{
    int i = blockIdx.x * 256 + threadIdx.x;
    if (i >= rows * Kpad) return;
    int r = i / Kpad, k = i - r * Kpad;
    float v = (k < Kin) ? s[(size_t)r * Kin + k] : 0.f;
    d[i] = f2bf(v);
}

__global__ void k_zero32(unsigned int* __restrict__ p, int n)
{
    int i = blockIdx.x * 256 + threadIdx.x;
    if (i < n) p[i] = 0u;
}

__global__ void k_embed(const int* __restrict__ fr, const float* __restrict__ emb,
                        u16* __restrict__ X)
{
    int i = blockIdx.x * 256 + threadIdx.x;
    if (i >= Bz * Sz * EPAD) return;
    int r = i / EPAD, k = i - r * EPAD;
    int tok = fr[r];
    float v = (k < Ee) ? emb[(size_t)tok * Ee + k] : 0.f;
    X[i] = f2bf(v);
}

// scores[b,s] = tanh(te[b,s,:] + qa[b,:]) . tot_w + tot_b
__global__ __launch_bounds__(256) void k_attn_score(
    const float* __restrict__ te, const float* __restrict__ qa,
    const float* __restrict__ tot_w, const float* __restrict__ tot_b,
    float* __restrict__ scores)
{
    int bs = blockIdx.x, b = bs >> 6, j = threadIdx.x;
    float v = tanhf(te[(size_t)bs * 256 + j] + qa[(size_t)b * 256 + j]);
    __shared__ float red[256];
    red[j] = v * tot_w[j];
    __syncthreads();
    for (int s = 128; s > 0; s >>= 1) {
        if (j < s) red[j] += red[j + s];
        __syncthreads();
    }
    if (j == 0) scores[bs] = red[0] + tot_b[0];
}

// softmax + weighted sum -> cat[b,0:512]; token embedding -> cat[b,512:832]
__global__ __launch_bounds__(256) void k_attn_ctx(
    const float* __restrict__ scores, const float* __restrict__ enc,
    const float* __restrict__ en_emb, const int* __restrict__ tok, int use_bos,
    u16* __restrict__ cat)
{
    int b = blockIdx.x, t = threadIdx.x;
    __shared__ float p[64];
    __shared__ float red[256];
    float sc = (t < 64) ? scores[(size_t)b * 64 + t] : -3.4e38f;
    red[t] = sc; __syncthreads();
    for (int s = 128; s > 0; s >>= 1) { if (t < s) red[t] = fmaxf(red[t], red[t + s]); __syncthreads(); }
    float mx = red[0]; __syncthreads();
    float e = (t < 64) ? __expf(sc - mx) : 0.f;
    red[t] = e; __syncthreads();
    for (int s = 128; s > 0; s >>= 1) { if (t < s) red[t] += red[t + s]; __syncthreads(); }
    float inv = 1.f / red[0];
    if (t < 64) p[t] = e * inv;
    __syncthreads();
    for (int c = t; c < 512; c += 256) {
        float acc = 0.f;
        for (int s = 0; s < 64; ++s)
            acc += p[s] * enc[((size_t)b * 64 + s) * 512 + c];
        cat[(size_t)b * CPAD + c] = f2bf(acc);
    }
    int token = use_bos ? 1 : tok[b];
    for (int k = t; k < 320; k += 256) {
        float v = (k < Ee) ? en_emb[(size_t)token * Ee + k] : 0.f;
        cat[(size_t)b * CPAD + 512 + k] = f2bf(v);
    }
}

__global__ __launch_bounds__(256) void k_argmax(
    const float* __restrict__ logits, int ld, int* __restrict__ tok)
{
    int b = blockIdx.x, t = threadIdx.x;
    const float* row = logits + (size_t)b * ld;
    float bestv = -3.4e38f; int besti = 0;
    for (int v = t; v < VEv; v += 256) {
        float x = row[v];
        if (x > bestv) { bestv = x; besti = v; }
    }
    __shared__ float sv[256]; __shared__ int si[256];
    sv[t] = bestv; si[t] = besti; __syncthreads();
    for (int s = 128; s > 0; s >>= 1) {
        if (t < s) {
            if (sv[t + s] > sv[t] || (sv[t + s] == sv[t] && si[t + s] < si[t])) {
                sv[t] = sv[t + s]; si[t] = si[t + s];
            }
        }
        __syncthreads();
    }
    if (t == 0) tok[b] = si[0];
}

// ---------------------------------------------------------------------------
extern "C" void kernel_launch(void* const* d_in, const int* in_sizes, int n_in,
                              void* d_out, int out_size, void* d_ws, size_t ws_size,
                              hipStream_t stream)
{
    (void)in_sizes; (void)n_in; (void)out_size; (void)ws_size;
    const int*   fr       = (const int*)  d_in[0];
    const float* fr_emb   = (const float*)d_in[1];
    const float* en_emb   = (const float*)d_in[2];
    const float* enc0_wih = (const float*)d_in[3];
    const float* enc0_whh = (const float*)d_in[4];
    const float* enc0_bih = (const float*)d_in[5];
    const float* enc0_bhh = (const float*)d_in[6];
    const float* enc1_wih = (const float*)d_in[7];
    const float* enc1_whh = (const float*)d_in[8];
    const float* enc1_bih = (const float*)d_in[9];
    const float* enc1_bhh = (const float*)d_in[10];
    const float* dec0_wih = (const float*)d_in[11];
    const float* dec0_whh = (const float*)d_in[12];
    const float* dec0_bih = (const float*)d_in[13];
    const float* dec0_bhh = (const float*)d_in[14];
    const float* dec1_wih = (const float*)d_in[15];
    const float* dec1_whh = (const float*)d_in[16];
    const float* dec1_bih = (const float*)d_in[17];
    const float* dec1_bhh = (const float*)d_in[18];
    const float* hid_w    = (const float*)d_in[19];
    const float* hid_b    = (const float*)d_in[20];
    const float* encT_w   = (const float*)d_in[21];
    const float* encT_b   = (const float*)d_in[22];
    const float* attn_w   = (const float*)d_in[23];
    const float* attn_b   = (const float*)d_in[24];
    const float* tot_w    = (const float*)d_in[25];
    const float* tot_b    = (const float*)d_in[26];
    const float* rnnT_w   = (const float*)d_in[27];
    const float* rnnT_b   = (const float*)d_in[28];
    const float* voc_w    = (const float*)d_in[29];
    const float* voc_b    = (const float*)d_in[30];
    float* out = (float*)d_out;

    const size_t SCAN_LDS = 64 * 768 * 4 + 64 * 256 * 4 + 64 * 256 * 2; // 294912
    const size_t GS_LDS   = 64 * 768 * 4;                               // 196608
    (void)hipFuncSetAttribute(reinterpret_cast<const void*>(k_scan),
                              hipFuncAttributeMaxDynamicSharedMemorySize, (int)SCAN_LDS);
    (void)hipFuncSetAttribute(reinterpret_cast<const void*>(k_grustep),
                              hipFuncAttributeMaxDynamicSharedMemorySize, (int)GS_LDS);

    // ---- workspace carve-out ----
    char* base = (char*)d_ws;
    size_t off = 0;
    auto alloc = [&](size_t elems, size_t esz) -> void* {
        void* p = base + off;
        off += (elems * esz + 255) & ~(size_t)255;
        return p;
    };
    u16* W_e0wih0 = (u16*)alloc(768 * EPAD, 2);
    u16* W_e0wih1 = (u16*)alloc(768 * EPAD, 2);
    u16* W_e0whh0 = (u16*)alloc(768 * 256, 2);
    u16* W_e0whh1 = (u16*)alloc(768 * 256, 2);
    u16* W_e1wih0 = (u16*)alloc(768 * 512, 2);
    u16* W_e1wih1 = (u16*)alloc(768 * 512, 2);
    u16* W_e1whh0 = (u16*)alloc(768 * 256, 2);
    u16* W_e1whh1 = (u16*)alloc(768 * 256, 2);
    u16* W_d0wih  = (u16*)alloc(768 * EPAD, 2);
    u16* W_d0whh  = (u16*)alloc(768 * 256, 2);
    u16* W_d1wih  = (u16*)alloc(768 * 256, 2);
    u16* W_d1whh  = (u16*)alloc(768 * 256, 2);
    u16* W_hid    = (u16*)alloc(256 * 512, 2);
    u16* W_encT   = (u16*)alloc(256 * 512, 2);
    u16* W_attn   = (u16*)alloc(256 * 256, 2);
    u16* W_rnnT   = (u16*)alloc(Ee * CPAD, 2);
    u16* W_voc    = (u16*)alloc((size_t)VEv * 256, 2);

    u16*   Xbf    = (u16*)  alloc((size_t)Bz * Sz * EPAD, 2);
    float* gi_f   = (float*)alloc((size_t)Bz * Sz * 768, 4);
    float* gi_b   = (float*)alloc((size_t)Bz * Sz * 768, 4);
    u16*   cat1   = (u16*)  alloc((size_t)Bz * Sz * 512, 2);   // layer1 input (bf16)
    float* enc    = (float*)alloc((size_t)Bz * Sz * 512, 4);   // encoding (f32)
    u16*   encbf  = (u16*)  alloc((size_t)Bz * Sz * 512, 2);
    float* hsbuf  = (float*)alloc(4 * (size_t)Bz * 256, 4);    // stacked final h's
    u16*   hcat   = (u16*)  alloc(128 * 512, 2);
    float* dech   = (float*)alloc(128 * 256, 4);               // [h0 ; h1]
    u16*   dechbf = (u16*)  alloc(128 * 256, 2);
    float* te     = (float*)alloc((size_t)Bz * Sz * 256, 4);   // trans_enc
    float* qa     = (float*)alloc((size_t)Bz * 256, 4);
    float* scores = (float*)alloc((size_t)Bz * Sz, 4);
    u16*   catbuf = (u16*)  alloc((size_t)Bz * CPAD, 2);       // [weighted|seq|pad]
    u16*   xtbf   = (u16*)  alloc((size_t)Bz * EPAD, 2);
    float* gidec  = (float*)alloc((size_t)Bz * 768, 4);
    int*   tokbuf = (int*)  alloc(Bz, 4);

    auto conv = [&](const float* s, u16* d, int rows, int Kin, int Kpad) {
        int total = rows * Kpad;
        k_cvt<<<(total + 255) / 256, 256, 0, stream>>>(s, d, rows, Kin, Kpad);
    };
    auto gemm = [&](const u16* A, int lda, const u16* W, int ldb,
                    float* C, int ldc, u16* Cb, int ldcb,
                    const float* bias, int M, int N, int K) {
        dim3 g((N + 255) / 256, M / 64);
        k_gemm4<<<g, 256, 0, stream>>>(A, lda, W, ldb, C, ldc, Cb, ldcb, bias, N, K);
    };

    // ---- weight conversion (f32 -> zero-padded bf16) ----
    conv(enc0_wih,             W_e0wih0, 768, Ee, EPAD);
    conv(enc0_wih + 768 * Ee,  W_e0wih1, 768, Ee, EPAD);
    conv(enc0_whh,             W_e0whh0, 768, 256, 256);
    conv(enc0_whh + 768 * 256, W_e0whh1, 768, 256, 256);
    conv(enc1_wih,             W_e1wih0, 768, 512, 512);
    conv(enc1_wih + 768 * 512, W_e1wih1, 768, 512, 512);
    conv(enc1_whh,             W_e1whh0, 768, 256, 256);
    conv(enc1_whh + 768 * 256, W_e1whh1, 768, 256, 256);
    conv(dec0_wih, W_d0wih, 768, Ee, EPAD);
    conv(dec0_whh, W_d0whh, 768, 256, 256);
    conv(dec1_wih, W_d1wih, 768, 256, 256);
    conv(dec1_whh, W_d1whh, 768, 256, 256);
    conv(hid_w,  W_hid,  256, 512, 512);
    conv(encT_w, W_encT, 256, 512, 512);
    conv(attn_w, W_attn, 256, 256, 256);
    conv(rnnT_w, W_rnnT, Ee, 812, CPAD);
    conv(voc_w,  W_voc,  VEv, 256, 256);

    // ---- embeddings, pad init ----
    k_embed<<<(Bz * Sz * EPAD + 255) / 256, 256, 0, stream>>>(fr, fr_emb, Xbf);
    k_zero32<<<(Bz * EPAD / 2 + 255) / 256, 256, 0, stream>>>((unsigned int*)xtbf, Bz * EPAD / 2);

    // ---- encoder ----
    // layer 0: gi = X @ wih^T + bih batched over all timesteps, then persistent scans
    gemm(Xbf, EPAD, W_e0wih0, EPAD, gi_f, 768, nullptr, 0, enc0_bih,       Bz * Sz, 768, EPAD);
    gemm(Xbf, EPAD, W_e0wih1, EPAD, gi_b, 768, nullptr, 0, enc0_bih + 768, Bz * Sz, 768, EPAD);
    k_scan<<<1, 1024, SCAN_LDS, stream>>>(gi_f, W_e0whh0, enc0_bhh,       nullptr, cat1,       hsbuf + 0 * 16384, 0);
    k_scan<<<1, 1024, SCAN_LDS, stream>>>(gi_b, W_e0whh1, enc0_bhh + 768, nullptr, cat1 + 256, hsbuf + 1 * 16384, 1);

    // layer 1: input = concat(yf, yb)
    gemm(cat1, 512, W_e1wih0, 512, gi_f, 768, nullptr, 0, enc1_bih,       Bz * Sz, 768, 512);
    gemm(cat1, 512, W_e1wih1, 512, gi_b, 768, nullptr, 0, enc1_bih + 768, Bz * Sz, 768, 512);
    k_scan<<<1, 1024, SCAN_LDS, stream>>>(gi_f, W_e1whh0, enc1_bhh,       enc,       encbf,       hsbuf + 2 * 16384, 0);
    k_scan<<<1, 1024, SCAN_LDS, stream>>>(gi_b, W_e1whh1, enc1_bhh + 768, enc + 256, encbf + 256, hsbuf + 3 * 16384, 1);

    // hidden = stack(hs).reshape(2,B,512) @ hid_w^T + hid_b  (flat reinterpret of hsbuf)
    conv(hsbuf, hcat, 128, 512, 512);
    gemm(hcat, 512, W_hid, 512, dech, 256, dechbf, 256, hid_b, 128, 256, 512);

    // trans_enc = encoding @ encT_w^T + encT_b
    gemm(encbf, 512, W_encT, 512, te, 256, nullptr, 0, encT_b, Bz * Sz, 256, 512);

    // ---- decoder ----
    for (int t = 0; t < Tz; ++t) {
        // qa = h1 @ attn_w^T + attn_b
        gemm(dechbf + 64 * 256, 256, W_attn, 256, qa, 256, nullptr, 0, attn_b, Bz, 256, 256);
        k_attn_score<<<Bz * Sz, 256, 0, stream>>>(te, qa, tot_w, tot_b, scores);
        k_attn_ctx<<<Bz, 256, 0, stream>>>(scores, enc, en_emb, tokbuf, (t == 0) ? 1 : 0, catbuf);

        // xt = [weighted|seq] @ rnnT_w^T + rnnT_b  (bf16 only, zero-padded cols)
        gemm(catbuf, CPAD, W_rnnT, CPAD, nullptr, 0, xtbf, EPAD, rnnT_b, Bz, Ee, CPAD);

        // dec GRU 0 (fused gh GEMM + pointwise)
        gemm(xtbf, EPAD, W_d0wih, EPAD, gidec, 768, nullptr, 0, dec0_bih, Bz, 768, EPAD);
        k_grustep<<<1, 1024, GS_LDS, stream>>>(gidec, dechbf, W_d0whh, dec0_bhh,
                                               dech, dech, dechbf);
        // dec GRU 1 (input = new h0, state = old h1)
        gemm(dechbf, 256, W_d1wih, 256, gidec, 768, nullptr, 0, dec1_bih, Bz, 768, 256);
        k_grustep<<<1, 1024, GS_LDS, stream>>>(gidec, dechbf + 64 * 256, W_d1whh, dec1_bhh,
                                               dech + 64 * 256, dech + 64 * 256, dechbf + 64 * 256);

        // logits[b, t, :] = h1 @ voc_w^T + voc_b   (written straight into d_out)
        gemm(dechbf + 64 * 256, 256, W_voc, 256, out + (size_t)t * VEv, Tz * VEv,
             nullptr, 0, voc_b, Bz, VEv, 256);
        k_argmax<<<Bz, 256, 0, stream>>>(out + (size_t)t * VEv, Tz * VEv, tokbuf);
    }
}
